// DyNet_10488310137286
// MI455X (gfx1250) — compile-verified, matrix operand order checked by reading, not run on hardware
//
#include <hip/hip_runtime.h>

typedef __attribute__((ext_vector_type(16))) __bf16 v16bf;
typedef __attribute__((ext_vector_type(8)))  __bf16 v8bf;
typedef __attribute__((ext_vector_type(8)))  float  v8f;

#define B_    16
#define CIN   128
#define COUT  128
#define Hh    64
#define Ww    64
#define KNUM  8
#define HO    62
#define WO    62
#define NPIX  (HO * WO)        /* 3844 */
#define KTOT  (CIN * 9)        /* 1152, tap-major: k = r*128 + ci */
#define NT    128              /* N tile per block   */
#define KC    32               /* K chunk = one WMMA k-step */
#define LDR   40               /* padded LDS row stride (bf16), 80B = 5*16B */
#define NSTAGE 36              /* 9 taps x 4 ci-chunks */

__device__ __forceinline__ unsigned short f2bf(float f) {
    union { float f; unsigned int u; } v; v.f = f;
    unsigned int u = v.u;
    unsigned int r = (u + 0x7FFFu + ((u >> 16) & 1u)) >> 16;  // RNE
    return (unsigned short)r;
}

// ---- stage 0: fp32 NCHW -> bf16 NHWC (LDS tile transpose, coalesced both sides)
__global__ void __launch_bounds__(256) cvt_nhwc_kernel(const float* __restrict__ in,
                                                       unsigned short* __restrict__ bfin) {
    __shared__ unsigned short sT[64][34];          // [pix][ci], padded
    const int b   = blockIdx.y;
    const int t   = blockIdx.x;                    // 64 pix-tiles x 4 ci-tiles
    const int p0  = (t & 63) * 64;
    const int ci0 = (t >> 6) * 32;
#pragma unroll
    for (int j = 0; j < 8; ++j) {                  // read 32ci x 64pix, coalesced in pix
        const int idx = threadIdx.x + j * 256;
        const int ci = idx >> 6, p = idx & 63;
        const float v = in[((size_t)(b * CIN + ci0 + ci)) * (Hh * Ww) + p0 + p];
        sT[p][ci] = f2bf(v);
    }
    __syncthreads();
#pragma unroll
    for (int j = 0; j < 8; ++j) {                  // write NHWC, coalesced in ci
        const int idx = threadIdx.x + j * 256;
        const int p = idx >> 5, ci = idx & 31;
        bfin[((size_t)b * (Hh * Ww) + p0 + p) * CIN + ci0 + ci] = sT[p][ci];
    }
}

// ---- stage 1: global average pool -> avg (B, Cin) -------------------------
__global__ void __launch_bounds__(256) avg_kernel(const float* __restrict__ in,
                                                  float* __restrict__ avg) {
    __shared__ float red[256];
    const int bc = blockIdx.x;
    const float* p = in + (size_t)bc * (Hh * Ww);
    float s = 0.f;
    for (int i = threadIdx.x; i < Hh * Ww; i += 256) s += p[i];
    red[threadIdx.x] = s;
    __syncthreads();
    for (int off = 128; off > 0; off >>= 1) {
        if (threadIdx.x < off) red[threadIdx.x] += red[threadIdx.x + off];
        __syncthreads();
    }
    if (threadIdx.x == 0) avg[bc] = red[0] * (1.0f / (Hh * Ww));
}

// ---- stage 2: gate = avg @ fc_w^T + fc_b (B, K) ---------------------------
__global__ void gate_kernel(const float* __restrict__ avg,
                            const float* __restrict__ fc_w,
                            const float* __restrict__ fc_b,
                            float* __restrict__ gate) {
    const int t = threadIdx.x;                     // 0..127
    const int b = t >> 3, k = t & 7;
    float s = fc_b[k];
    for (int ci = 0; ci < CIN; ++ci) s += avg[b * CIN + ci] * fc_w[k * CIN + ci];
    gate[b * KNUM + k] = s;
}

// ---- stage 3: dyn[b][co][r*128+ci] bf16 (tap-major) -----------------------
__global__ void __launch_bounds__(256) dyn_kernel(const float* __restrict__ gate,
                                                  const float* __restrict__ weight,
                                                  unsigned short* __restrict__ dynbf) {
    const int b   = blockIdx.y;
    const int idx = blockIdx.x * 256 + threadIdx.x;     // co*1152 + ci*9 + r
    if (idx >= COUT * KTOT) return;
    float s = 0.f;
#pragma unroll
    for (int k = 0; k < KNUM; ++k)
        s += gate[b * KNUM + k] * weight[(size_t)k * COUT * KTOT + idx];
    const int co  = idx / KTOT;
    const int rem = idx - co * KTOT;
    const int ci  = rem / 9;
    const int r   = rem - ci * 9;
    dynbf[((size_t)b * COUT + co) * KTOT + r * CIN + ci] = f2bf(s);
}

// ---- stage 4: implicit-GEMM conv, 128x128 block tile, waves 4Mx2N (32x64 each)
__global__ void __launch_bounds__(256) conv_wmma_kernel(
        const unsigned short* __restrict__ bfin,   // (B, H, W, Cin) bf16 NHWC
        const unsigned short* __restrict__ dynbf,  // (B, Cout, 1152) tap-major bf16
        float* __restrict__ out) {
    __shared__ unsigned short sA[2 * COUT * LDR];  // ping-pong 128x32 A tiles
    __shared__ unsigned short sB[2 * NT * LDR];    // ping-pong 128x32 B tiles

    const int tid  = threadIdx.x;
    const int b    = blockIdx.y;
    const int n0   = blockIdx.x * NT;
    const int wave = tid >> 5;
    const int lane = tid & 31;
    const int half = lane >> 4;
    const int lm   = lane & 15;
    const int koff = half * 8;
    const int mw   = (wave >> 1) * 32;             // wave's M origin (4 rows of waves)
    const int nwv  = (wave & 1) * 64;              // wave's N origin (2 cols of waves)

    // loop-invariant im2col coordinates (thread always fills pixel nn)
    const int nn    = tid & 127;
    const int kbase = (tid >> 7) * 16;             // 0 or 16 (ci sub-chunk)
    const int n     = n0 + nn;
    const bool nval = (n < NPIX);
    const int y     = n / WO;
    const int x     = n - y * WO;
    const int pix0  = y * Ww + x;
    const unsigned short* inB = bfin + (size_t)b * (Hh * Ww) * CIN;

    v8f acc[2][4];
#pragma unroll
    for (int am = 0; am < 2; ++am)
#pragma unroll
        for (int t = 0; t < 4; ++t) acc[am][t] = v8f{};

    const unsigned short* Abase = dynbf + (size_t)b * COUT * KTOT;

    uint4 ra[2];
    uint4 rb[2];

    auto load_regs = [&](int s) {
        const int ci0 = (s & 3) * KC;
        const int r   = s >> 2;
        const int ky  = (r * 171) >> 9;            // r/3 for r<9
        const int kx  = r - 3 * ky;
        const int k0  = r * CIN + ci0;
#pragma unroll
        for (int j = 0; j < 2; ++j) {
            const int i  = tid + j * 256;          // < 512
            const int m  = i >> 2;
            const int kq = (i & 3) * 8;
            ra[j] = *(const uint4*)(Abase + (size_t)m * KTOT + k0 + kq);
        }
        const unsigned short* src =
            inB + (size_t)(pix0 + ky * Ww + kx) * CIN + ci0 + kbase;
        if (nval) {
            rb[0] = *(const uint4*)(src);
            rb[1] = *(const uint4*)(src + 8);
        } else {
            rb[0] = make_uint4(0, 0, 0, 0);
            rb[1] = make_uint4(0, 0, 0, 0);
        }
    };

    auto store_lds = [&](int buf) {
        unsigned short* pA = sA + buf * (COUT * LDR);
        unsigned short* pB = sB + buf * (NT * LDR);
#pragma unroll
        for (int j = 0; j < 2; ++j) {
            const int i  = tid + j * 256;
            const int m  = i >> 2;
            const int kq = (i & 3) * 8;
            *(uint4*)&pA[m * LDR + kq] = ra[j];
        }
        *(uint4*)&pB[nn * LDR + kbase]     = rb[0];
        *(uint4*)&pB[nn * LDR + kbase + 8] = rb[1];
    };

    auto compute = [&](int buf) {
        const unsigned short* pA = sA + buf * (COUT * LDR);
        const unsigned short* pB = sB + buf * (NT * LDR);
        v16bf a[2];
#pragma unroll
        for (int am = 0; am < 2; ++am) {
            v8bf alo = *(const v8bf*)&pA[(mw + am * 16 + lm) * LDR + koff];
            v8bf ahi = *(const v8bf*)&pA[(mw + am * 16 + lm) * LDR + koff + 16];
            a[am] = __builtin_shufflevector(alo, ahi,
                    0, 1, 2, 3, 4, 5, 6, 7, 8, 9, 10, 11, 12, 13, 14, 15);
        }
#pragma unroll
        for (int t = 0; t < 4; ++t) {
            v8bf blo = *(const v8bf*)&pB[(nwv + t * 16 + lm) * LDR + koff];
            v8bf bhi = *(const v8bf*)&pB[(nwv + t * 16 + lm) * LDR + koff + 16];
            v16bf bv = __builtin_shufflevector(blo, bhi,
                      0, 1, 2, 3, 4, 5, 6, 7, 8, 9, 10, 11, 12, 13, 14, 15);
#pragma unroll
            for (int am = 0; am < 2; ++am)
                acc[am][t] = __builtin_amdgcn_wmma_f32_16x16x32_bf16(
                                 false, a[am], false, bv, (short)0, acc[am][t],
                                 false, false);
        }
    };

    // software pipeline: one barrier per stage, global latency behind compute
    load_regs(0);
    store_lds(0);
    __syncthreads();
    for (int s = 0; s < NSTAGE; ++s) {
        const int cur = s & 1;
        if (s + 1 < NSTAGE) load_regs(s + 1);
        compute(cur);
        if (s + 1 < NSTAGE) {
            store_lds((s + 1) & 1);
            __syncthreads();
        }
    }

    // D layout: VGPR j -> M = j + 8*half, N = lm
    float* outB = out + (size_t)b * COUT * NPIX;
#pragma unroll
    for (int am = 0; am < 2; ++am) {
        const int mbase = mw + am * 16 + half * 8;
#pragma unroll
        for (int t = 0; t < 4; ++t) {
            const int ng = n0 + nwv + t * 16 + lm;
            if (ng < NPIX) {
#pragma unroll
                for (int j = 0; j < 8; ++j)
                    outB[(size_t)(mbase + j) * NPIX + ng] = acc[am][t][j];
            }
        }
    }
}

extern "C" void kernel_launch(void* const* d_in, const int* in_sizes, int n_in,
                              void* d_out, int out_size, void* d_ws, size_t ws_size,
                              hipStream_t stream) {
    (void)in_sizes; (void)n_in; (void)out_size; (void)ws_size;
    const float* in   = (const float*)d_in[0];
    const float* wgt  = (const float*)d_in[1];
    const float* fc_w = (const float*)d_in[2];
    const float* fc_b = (const float*)d_in[3];
    float* out = (float*)d_out;

    // ws layout: gate(128 f32) | avg(2048 f32) | dynbf(2.36M bf16) | bfin(8.4M bf16 NHWC)
    float* gate = (float*)d_ws;
    float* avg  = gate + 128;
    unsigned short* dynbf = (unsigned short*)(avg + 2048);               // +8704 B
    unsigned short* bfin  = dynbf + (size_t)B_ * COUT * KTOT;            // +4718592 B

    cvt_nhwc_kernel<<<dim3(256, B_), 256, 0, stream>>>(in, bfin);
    avg_kernel <<<B_ * CIN, 256, 0, stream>>>(in, avg);
    gate_kernel<<<1, 128, 0, stream>>>(avg, fc_w, fc_b, gate);
    dyn_kernel <<<dim3((COUT * KTOT) / 256, B_), 256, 0, stream>>>(gate, wgt, dynbf);
    conv_wmma_kernel<<<dim3((NPIX + NT - 1) / NT, B_), 256, 0, stream>>>(bfin, dynbf, out);
}
